// TwoPhaseMoE_26362509263287
// MI455X (gfx1250) — matrix operand order repaired and checked by reading, not compiled, not run on hardware
//
#include <hip/hip_runtime.h>
#include <hip/hip_bf16.h>
#include <math.h>

// Problem constants (from reference): B=2,S=2048 -> N=4096 tokens
#define N_TOK 4096
#define D_DIM 1024
#define F_DIM 2048
#define E_NUM 14
#define M_TILE 32

typedef __attribute__((ext_vector_type(8)))  __bf16 v8bf;
typedef __attribute__((ext_vector_type(16))) __bf16 v16bf;
typedef __attribute__((ext_vector_type(8)))  float  v8f;

// ---------------- LDS layout for moe kernel (dynamic) ----------------
// Xs : 32 x (1024+8) bf16      =  66048 B  (A tiles for GEMM1, row-major, padded)
// Hs : 32 x (2048+8) bf16      = 131584 B  (gelu(XW1+b1) bf16, A tiles for GEMM2)
// Bs : 2  x 128 x 40 bf16      =  20480 B  (double-buffered B tiles, column-major K)
// tok: 32 int, wts: 32 float
#define XS_STRIDE  (D_DIM + 8)
#define HS_STRIDE  (F_DIM + 8)
#define BS_STRIDE  40
#define OFF_XS     0
#define OFF_HS     (OFF_XS + M_TILE * XS_STRIDE * 2)        // 66048
#define OFF_BS     (OFF_HS + M_TILE * HS_STRIDE * 2)        // 197632
#define OFF_TOK    (OFF_BS + 2 * 128 * BS_STRIDE * 2)       // 218112
#define OFF_WT     (OFF_TOK + M_TILE * 4)                   // 218240
#define SMEM_BYTES (OFF_WT + M_TILE * 4)                    // 218368 (< 320KB WGP LDS)

// ---------------- fragment loaders (per ISA 7.12.2 layouts, wave32) ----------------
// A 16x32 bf16: lane L holds row M=L&15; K chunks {K0..K0+7, K0+16..K0+23}, K0=(L<16?0:8)
__device__ __forceinline__ v16bf load_frag_A(const __bf16* row_k, int aK0) {
    v8bf lo = *(const v8bf*)(row_k + aK0);
    v8bf hi = *(const v8bf*)(row_k + aK0 + 16);
    return __builtin_shufflevector(lo, hi, 0,1,2,3,4,5,6,7,8,9,10,11,12,13,14,15);
}
// B 32x16 bf16: lane L holds col N=L&15; K chunks {K0..K0+7, K0+16..K0+23}, K0=(L<16?0:8)
__device__ __forceinline__ v16bf load_frag_B(const __bf16* col_k, int bK0) {
    v8bf lo = *(const v8bf*)(col_k + bK0);
    v8bf hi = *(const v8bf*)(col_k + bK0 + 16);
    return __builtin_shufflevector(lo, hi, 0,1,2,3,4,5,6,7,8,9,10,11,12,13,14,15);
}

// ---------------- kernel 0: zero per-expert counters ----------------
__global__ void moe_init_kernel(int* __restrict__ counts) {
    if (threadIdx.x < E_NUM) counts[threadIdx.x] = 0;
}

// ---------------- kernel 1: router (one wave per token) ----------------
__global__ void moe_router_kernel(const float* __restrict__ x,
                                  const float* __restrict__ rw,
                                  int* __restrict__ assign,
                                  float* __restrict__ wt) {
    const int wave = threadIdx.x >> 5;
    const int lane = threadIdx.x & 31;
    const int t = blockIdx.x * 8 + wave;           // grid = 512 blocks x 256 thr
    float acc[E_NUM];
#pragma unroll
    for (int e = 0; e < E_NUM; ++e) acc[e] = 0.0f;
    const float* xr = x + (size_t)t * D_DIM;
    for (int j = 0; j < D_DIM / 32; ++j) {
        const int d = j * 32 + lane;
        const float xv = xr[d];
        const float* rwd = rw + (size_t)d * E_NUM;
#pragma unroll
        for (int e = 0; e < E_NUM; ++e) acc[e] += xv * rwd[e];
    }
#pragma unroll
    for (int off = 16; off > 0; off >>= 1) {
#pragma unroll
        for (int e = 0; e < E_NUM; ++e) acc[e] += __shfl_xor(acc[e], off, 32);
    }
    if (lane == 0) {
        float m = acc[0];
        int amax = 0;
#pragma unroll
        for (int e = 1; e < E_NUM; ++e) {
            if (acc[e] > m) { m = acc[e]; amax = e; }   // first-max (jnp.argmax)
        }
        float s = 0.0f;
#pragma unroll
        for (int e = 0; e < E_NUM; ++e) s += expf(acc[e] - m);
        assign[t] = amax;
        wt[t] = 1.0f / s;                                // exp(0)/sum
    }
}

// ---------------- kernel 2: bucket tokens per expert ----------------
__global__ void moe_scatter_kernel(const int* __restrict__ assign,
                                   int* __restrict__ counts,
                                   int* __restrict__ lists) {
    const int t = blockIdx.x * blockDim.x + threadIdx.x;
    if (t < N_TOK) {
        const int e = assign[t];
        const int pos = atomicAdd(&counts[e], 1);
        lists[e * N_TOK + pos] = t;
    }
}

// ---------------- kernel 3: expert FFN, bf16 WMMA, fp32 accumulate ----------------
// grid = E_NUM * (N_TOK/M_TILE) blocks; block = 256 threads (8 waves); tile = 32 tokens
__global__ void __launch_bounds__(256)
moe_ffn_kernel(const int* __restrict__ lists, const int* __restrict__ counts,
               const float* __restrict__ wt, const float* __restrict__ x,
               const float* __restrict__ w1, const float* __restrict__ b1,
               const float* __restrict__ w2, const float* __restrict__ b2,
               float* __restrict__ out) {
    extern __shared__ char smem[];
    __bf16* Xs = (__bf16*)(smem + OFF_XS);
    __bf16* Hs = (__bf16*)(smem + OFF_HS);
    __bf16* Bs = (__bf16*)(smem + OFF_BS);
    int*    tokLds = (int*)(smem + OFF_TOK);
    float*  wLds   = (float*)(smem + OFF_WT);

    const int e = blockIdx.x >> 7;                   // 128 tiles per expert
    const int m = blockIdx.x & 127;
    const int cnt = counts[e];
    if (m * M_TILE >= cnt) return;                   // uniform early exit

    const int tid = threadIdx.x;
    if (tid < M_TILE) {
        const int r = m * M_TILE + tid;
        const int t = (r < cnt) ? lists[e * N_TOK + r] : -1;
        tokLds[tid] = t;
        wLds[tid] = (t >= 0) ? wt[t] : 0.0f;
    }
    __syncthreads();

    // stage X tile (32 x 1024 fp32 -> bf16 LDS), float4-vectorized
#pragma unroll 4
    for (int j = 0; j < (M_TILE * D_DIM / 4) / 256; ++j) {
        const int idx = tid + j * 256;
        const int row = idx >> 8, col = (idx & 255) * 4;
        const int t = tokLds[row];
        float4 v = make_float4(0.f, 0.f, 0.f, 0.f);
        if (t >= 0) v = *(const float4*)(x + (size_t)t * D_DIM + col);
        __bf16* dst = Xs + row * XS_STRIDE + col;
        dst[0] = (__bf16)v.x; dst[1] = (__bf16)v.y;
        dst[2] = (__bf16)v.z; dst[3] = (__bf16)v.w;
    }
    __syncthreads();

    const float* w1e = w1 + (size_t)e * D_DIM * F_DIM;
    const float* w2e = w2 + (size_t)e * F_DIM * D_DIM;
    const float* b1e = b1 + (size_t)e * F_DIM;
    const float* b2e = b2 + (size_t)e * D_DIM;

    const int wave = tid >> 5;
    const int lane = tid & 31;
    const int hi = lane >> 4;       // 0 or 1
    const int ln = lane & 15;
    const int aK0 = hi ? 8 : 0;
    const int bK0 = hi ? 8 : 0;

    // ---- GEMM1: H = gelu(X @ W1 + b1), 32 x 2048, K = 1024 ----
    for (int nb = 0; nb < F_DIM / 128; ++nb) {
        const int colbase = nb * 128;
        v8f acc0 = {};
        v8f acc1 = {};
        for (int kk = 0; kk < D_DIM / 32; ++kk) {
            const int kbase = kk * 32;
            __bf16* bs = Bs + (kk & 1) * (128 * BS_STRIDE);
#pragma unroll
            for (int j = 0; j < 4; ++j) {            // stage 32x128 fp32 -> bf16 col-major
                const int idx = tid + j * 256;
                const int row = idx >> 5, col = (idx & 31) * 4;
                const float4 v = *(const float4*)(w1e + (size_t)(kbase + row) * F_DIM
                                                  + colbase + col);
                bs[(col + 0) * BS_STRIDE + row] = (__bf16)v.x;
                bs[(col + 1) * BS_STRIDE + row] = (__bf16)v.y;
                bs[(col + 2) * BS_STRIDE + row] = (__bf16)v.z;
                bs[(col + 3) * BS_STRIDE + row] = (__bf16)v.w;
            }
            __syncthreads();
            v16bf b  = load_frag_B(bs + (wave * 16 + ln) * BS_STRIDE, bK0);
            v16bf a0 = load_frag_A(Xs + ln * XS_STRIDE + kbase, aK0);
            v16bf a1 = load_frag_A(Xs + (16 + ln) * XS_STRIDE + kbase, aK0);
            acc0 = __builtin_amdgcn_wmma_f32_16x16x32_bf16(false, a0, false, b,
                                                           (short)0, acc0, false, false);
            acc1 = __builtin_amdgcn_wmma_f32_16x16x32_bf16(false, a1, false, b,
                                                           (short)0, acc1, false, false);
        }
        const int col = colbase + wave * 16 + ln;
        const float bias = b1e[col];
#pragma unroll
        for (int r = 0; r < 8; ++r) {                // C layout: M = r + 8*hi, N = ln
            float v = acc0[r] + bias;
            float g = 0.5f * v * (1.0f + erff(v * 0.70710678118654752f));
            Hs[(r + 8 * hi) * HS_STRIDE + col] = (__bf16)g;
            v = acc1[r] + bias;
            g = 0.5f * v * (1.0f + erff(v * 0.70710678118654752f));
            Hs[(16 + r + 8 * hi) * HS_STRIDE + col] = (__bf16)g;
        }
    }
    __syncthreads();

    // ---- GEMM2: Out = H @ W2 + b2 (scaled by router weight), K = 2048 ----
    for (int nb = 0; nb < D_DIM / 128; ++nb) {
        const int colbase = nb * 128;
        v8f acc0 = {};
        v8f acc1 = {};
        for (int kk = 0; kk < F_DIM / 32; ++kk) {
            const int kbase = kk * 32;
            __bf16* bs = Bs + (kk & 1) * (128 * BS_STRIDE);
#pragma unroll
            for (int j = 0; j < 4; ++j) {
                const int idx = tid + j * 256;
                const int row = idx >> 5, col = (idx & 31) * 4;
                const float4 v = *(const float4*)(w2e + (size_t)(kbase + row) * D_DIM
                                                  + colbase + col);
                bs[(col + 0) * BS_STRIDE + row] = (__bf16)v.x;
                bs[(col + 1) * BS_STRIDE + row] = (__bf16)v.y;
                bs[(col + 2) * BS_STRIDE + row] = (__bf16)v.z;
                bs[(col + 3) * BS_STRIDE + row] = (__bf16)v.w;
            }
            __syncthreads();
            v16bf b  = load_frag_B(bs + (wave * 16 + ln) * BS_STRIDE, bK0);
            v16bf a0 = load_frag_A(Hs + ln * HS_STRIDE + kbase, aK0);
            v16bf a1 = load_frag_A(Hs + (16 + ln) * HS_STRIDE + kbase, aK0);
            acc0 = __builtin_amdgcn_wmma_f32_16x16x32_bf16(false, a0, false, b,
                                                           (short)0, acc0, false, false);
            acc1 = __builtin_amdgcn_wmma_f32_16x16x32_bf16(false, a1, false, b,
                                                           (short)0, acc1, false, false);
        }
        const int col = colbase + wave * 16 + ln;
        const float bias = b2e[col];
#pragma unroll
        for (int r = 0; r < 8; ++r) {
            int row = r + 8 * hi;
            int t = tokLds[row];
            if (t >= 0) out[(size_t)t * D_DIM + col] = acc0[r] * wLds[row] + bias;
            row += 16;
            t = tokLds[row];
            if (t >= 0) out[(size_t)t * D_DIM + col] = acc1[r] * wLds[row] + bias;
        }
    }
}

// ---------------- launch ----------------
extern "C" void kernel_launch(void* const* d_in, const int* in_sizes, int n_in,
                              void* d_out, int out_size, void* d_ws, size_t ws_size,
                              hipStream_t stream) {
    const float* x  = (const float*)d_in[0];   // h  (4096,1024)
    const float* rw = (const float*)d_in[1];   // router_w (1024,14)
    const float* w1 = (const float*)d_in[2];   // (14,1024,2048)
    const float* b1 = (const float*)d_in[3];   // (14,2048)
    const float* w2 = (const float*)d_in[4];   // (14,2048,1024)
    const float* b2 = (const float*)d_in[5];   // (14,1024)
    float* out = (float*)d_out;

    char* ws = (char*)d_ws;
    int*   assign = (int*)(ws + 0);                        // 4096 ints
    float* wt     = (float*)(ws + 16384);                  // 4096 floats
    int*   counts = (int*)(ws + 32768);                    // 14 ints
    int*   lists  = (int*)(ws + 32832);                    // 14*4096 ints

    moe_init_kernel<<<1, 64, 0, stream>>>(counts);
    moe_router_kernel<<<N_TOK / 8, 256, 0, stream>>>(x, rw, assign, wt);
    moe_scatter_kernel<<<N_TOK / 256, 256, 0, stream>>>(assign, counts, lists);
    moe_ffn_kernel<<<E_NUM * (N_TOK / M_TILE), 256, SMEM_BYTES, stream>>>(
        lists, counts, wt, x, w1, b1, w2, b2, out);
}
